// DropoutMM_20332375179588
// MI455X (gfx1250) — matrix operand order repaired and checked by reading, not compiled, not run on hardware
//
#include <hip/hip_runtime.h>

// DropoutMM: out = (mask ? x/(1-p) : 0) @ W^T
// M = T = 8192, K = D_in = 2048, N = D_out = 8192, fp32 in/out.
// Compute-bound -> bf16 WMMA with hi/lo split (Markidis) for near-fp32 accuracy.
// Pass 1 splits x (with dropout) and W into bf16 hi/lo arrays; pass 2 is a
// double-buffered bf16 WMMA GEMM (3 WMMAs per logical fp32 tile), using
// GLOBAL_LOAD_ASYNC_TO_LDS (ASYNCcnt) for global->LDS when available.

#define T_DIM 8192
#define K_DIM 2048
#define N_DIM 8192
#define LDSK  48   // padded LDS row stride (bf16 elems): 96B, 16B-aligned

typedef __attribute__((ext_vector_type(16))) __bf16 v16bf;
typedef __attribute__((ext_vector_type(8)))  __bf16 v8bf;
typedef __attribute__((ext_vector_type(8)))  float  v8f;
typedef __attribute__((ext_vector_type(8)))  unsigned short ushort8;
typedef int v4i_g __attribute__((vector_size(16)));   // matches builtin's V4i

#if __has_builtin(__builtin_amdgcn_global_load_async_to_lds_b128)
#define HAVE_ASYNC_LDS 1
#else
#define HAVE_ASYNC_LDS 0
#endif

__device__ __forceinline__ unsigned short f32_to_bf16_rne(float f) {
    unsigned int u = __float_as_uint(f);
    unsigned int r = u + 0x7FFFu + ((u >> 16) & 1u);   // round-to-nearest-even
    return (unsigned short)(r >> 16);
}
__device__ __forceinline__ float bf16u_to_f32(unsigned short h) {
    return __uint_as_float(((unsigned int)h) << 16);
}

// copy 16 bytes global -> LDS, per lane
__device__ __forceinline__ void copy16_g2l(const unsigned short* g, unsigned short* l) {
#if HAVE_ASYNC_LDS
    __builtin_amdgcn_global_load_async_to_lds_b128(
        (__attribute__((address_space(1))) v4i_g*)(void*)const_cast<unsigned short*>(g),
        (__attribute__((address_space(3))) v4i_g*)(void*)l,
        0, 0);
#else
    *reinterpret_cast<ushort8*>(l) = *reinterpret_cast<const ushort8*>(g);
#endif
}

__device__ __forceinline__ void wait_async_zero() {
#if HAVE_ASYNC_LDS
 #if __has_builtin(__builtin_amdgcn_s_wait_asynccnt)
    __builtin_amdgcn_s_wait_asynccnt(0);
 #else
    asm volatile("s_wait_asynccnt 0x0" ::: "memory");
 #endif
#endif
}

// A-frag (16x32 bf16, ISA 7.12.2): lanes 0-15 -> M=lane, K = 0..7 & 16..23;
// lanes 16-31 -> M=lane-16, K = 8..15 & 24..31.
__device__ __forceinline__ v16bf load_fragA(const unsigned short (*tile)[LDSK],
                                            int rowBase, int lane) {
    const int r  = rowBase + (lane & 15);
    const int k0 = (lane >> 4) * 8;
    v8bf a = *reinterpret_cast<const v8bf*>(&tile[r][k0]);
    v8bf b = *reinterpret_cast<const v8bf*>(&tile[r][k0 + 16]);
    return __builtin_shufflevector(a, b, 0,1,2,3,4,5,6,7,8,9,10,11,12,13,14,15);
}

// B-frag (32x16 bf16): lanes 0-15 -> N=lane, K = 0..7 & 16..23; lanes 16-31 -> N=lane-16,
// K = 8..15 & 24..31. Tile stored [N][K] (W's native row-major layout).
__device__ __forceinline__ v16bf load_fragB(const unsigned short (*tile)[LDSK],
                                            int rowBase, int lane) {
    const int r  = rowBase + (lane & 15);
    const int k0 = (lane >> 4) * 8;
    v8bf a = *reinterpret_cast<const v8bf*>(&tile[r][k0]);
    v8bf b = *reinterpret_cast<const v8bf*>(&tile[r][k0 + 16]);
    return __builtin_shufflevector(a, b, 0,1,2,3,4,5,6,7,8,9,10,11,12,13,14,15);
}

// ---------------- Pass 1: elementwise split kernels ----------------

__global__ __launch_bounds__(256)
void split_x_dropout(const float* __restrict__ x,
                     const unsigned char* __restrict__ mask,
                     ushort8* __restrict__ hi, ushort8* __restrict__ lo) {
    const size_t i   = (size_t)blockIdx.x * blockDim.x + threadIdx.x;
    const size_t off = i * 8;
    const float scale = 1.0f / 0.9f;
    float4 a0 = *reinterpret_cast<const float4*>(x + off);
    float4 a1 = *reinterpret_cast<const float4*>(x + off + 4);
    unsigned long long mm =
        *reinterpret_cast<const unsigned long long*>(mask + off);
    float av[8] = {a0.x, a0.y, a0.z, a0.w, a1.x, a1.y, a1.z, a1.w};
    ushort8 h, l;
    #pragma unroll
    for (int e = 0; e < 8; ++e) {
        float f = ((mm >> (8 * e)) & 0xFFull) ? av[e] * scale : 0.0f;
        unsigned short hb = f32_to_bf16_rne(f);
        h[e] = hb;
        l[e] = f32_to_bf16_rne(f - bf16u_to_f32(hb));
    }
    hi[i] = h;
    lo[i] = l;
}

__global__ __launch_bounds__(256)
void split_w(const float* __restrict__ W,
             ushort8* __restrict__ hi, ushort8* __restrict__ lo) {
    const size_t i   = (size_t)blockIdx.x * blockDim.x + threadIdx.x;
    const size_t off = i * 8;
    float4 a0 = *reinterpret_cast<const float4*>(W + off);
    float4 a1 = *reinterpret_cast<const float4*>(W + off + 4);
    float av[8] = {a0.x, a0.y, a0.z, a0.w, a1.x, a1.y, a1.z, a1.w};
    ushort8 h, l;
    #pragma unroll
    for (int e = 0; e < 8; ++e) {
        float f = av[e];
        unsigned short hb = f32_to_bf16_rne(f);
        h[e] = hb;
        l[e] = f32_to_bf16_rne(f - bf16u_to_f32(hb));
    }
    hi[i] = h;
    lo[i] = l;
}

// ---------------- Pass 2: bf16 WMMA GEMM (pre-split inputs) ----------------
// 256 threads = 8 waves. Block tile 128(M) x 128(N), BK = 32, double-buffered.
// Wave grid 4(m) x 2(n): each wave owns 32x64 = 2x4 WMMA accumulators.

__global__ __launch_bounds__(256)
void gemm_bf16split(const unsigned short* __restrict__ gAhi,
                    const unsigned short* __restrict__ gAlo,
                    const unsigned short* __restrict__ gBhi,
                    const unsigned short* __restrict__ gBlo,
                    float* __restrict__ out) {
    __shared__ __align__(16) unsigned short sAhi[2][128][LDSK];
    __shared__ __align__(16) unsigned short sAlo[2][128][LDSK];
    __shared__ __align__(16) unsigned short sBhi[2][128][LDSK];
    __shared__ __align__(16) unsigned short sBlo[2][128][LDSK];

    const int tid  = threadIdx.x;
    const int lane = tid & 31;
    const int wave = tid >> 5;     // 0..7
    const int wm   = wave >> 1;    // 0..3 -> 32-row slice
    const int wn   = wave & 1;     // 0..1 -> 64-col slice

    const int mBase = blockIdx.y * 128;
    const int nBase = blockIdx.x * 128;

    // loader: 256 threads cover 128 rows x 32 cols bf16, 16B x2 per thread/array
    const int ldr = tid >> 1;          // 0..127
    const int ldc = (tid & 1) * 16;    // 0 or 16

    const unsigned short* gA0 = gAhi + (size_t)(mBase + ldr) * K_DIM + ldc;
    const unsigned short* gA1 = gAlo + (size_t)(mBase + ldr) * K_DIM + ldc;
    const unsigned short* gB0 = gBhi + (size_t)(nBase + ldr) * K_DIM + ldc;
    const unsigned short* gB1 = gBlo + (size_t)(nBase + ldr) * K_DIM + ldc;

    v8f acc[2][4];
    #pragma unroll
    for (int i = 0; i < 2; ++i)
        #pragma unroll
        for (int j = 0; j < 4; ++j)
            acc[i][j] = v8f{};

    const int NS = K_DIM / 32;   // 64 stages

    // prologue: fill buffer 0
    {
        copy16_g2l(gA0,     &sAhi[0][ldr][ldc]);
        copy16_g2l(gA0 + 8, &sAhi[0][ldr][ldc + 8]);
        copy16_g2l(gA1,     &sAlo[0][ldr][ldc]);
        copy16_g2l(gA1 + 8, &sAlo[0][ldr][ldc + 8]);
        copy16_g2l(gB0,     &sBhi[0][ldr][ldc]);
        copy16_g2l(gB0 + 8, &sBhi[0][ldr][ldc + 8]);
        copy16_g2l(gB1,     &sBlo[0][ldr][ldc]);
        copy16_g2l(gB1 + 8, &sBlo[0][ldr][ldc + 8]);
        wait_async_zero();
    }
    __syncthreads();

    for (int s = 0; s < NS; ++s) {
        const int cur = s & 1;

        // issue next stage's copies into the other buffer (under the WMMAs)
        if (s + 1 < NS) {
            const int nxt = cur ^ 1;
            const int ko  = (s + 1) * 32;
            copy16_g2l(gA0 + ko,     &sAhi[nxt][ldr][ldc]);
            copy16_g2l(gA0 + ko + 8, &sAhi[nxt][ldr][ldc + 8]);
            copy16_g2l(gA1 + ko,     &sAlo[nxt][ldr][ldc]);
            copy16_g2l(gA1 + ko + 8, &sAlo[nxt][ldr][ldc + 8]);
            copy16_g2l(gB0 + ko,     &sBhi[nxt][ldr][ldc]);
            copy16_g2l(gB0 + ko + 8, &sBhi[nxt][ldr][ldc + 8]);
            copy16_g2l(gB1 + ko,     &sBlo[nxt][ldr][ldc]);
            copy16_g2l(gB1 + ko + 8, &sBlo[nxt][ldr][ldc + 8]);
        }

        v16bf aH[2], aL[2], bH[4], bL[4];
        #pragma unroll
        for (int i = 0; i < 2; ++i) {
            aH[i] = load_fragA(sAhi[cur], wm * 32 + i * 16, lane);
            aL[i] = load_fragA(sAlo[cur], wm * 32 + i * 16, lane);
        }
        #pragma unroll
        for (int j = 0; j < 4; ++j) {
            bH[j] = load_fragB(sBhi[cur], wn * 64 + j * 16, lane);
            bL[j] = load_fragB(sBlo[cur], wn * 64 + j * 16, lane);
        }

        // split-bf16 fp32 emulation: hi*hi + hi*lo + lo*hi, fp32 accumulate
        #pragma unroll
        for (int i = 0; i < 2; ++i) {
            #pragma unroll
            for (int j = 0; j < 4; ++j) {
                v8f c = acc[i][j];
                c = __builtin_amdgcn_wmma_f32_16x16x32_bf16(
                        false, aL[i], false, bH[j], (short)0, c, false, false);
                c = __builtin_amdgcn_wmma_f32_16x16x32_bf16(
                        false, aH[i], false, bL[j], (short)0, c, false, false);
                c = __builtin_amdgcn_wmma_f32_16x16x32_bf16(
                        false, aH[i], false, bH[j], (short)0, c, false, false);
                acc[i][j] = c;
            }
        }

        wait_async_zero();   // next buffer fully written (async path)
        __syncthreads();     // all waves done reading current buffer
    }

    // C tile layout: lanes 0-15: N=lane, M=vgpr; lanes 16-31: N=lane-16, M=8+vgpr
    const int cn = lane & 15;
    const int mh = (lane >> 4) * 8;
    #pragma unroll
    for (int i = 0; i < 2; ++i) {
        #pragma unroll
        for (int j = 0; j < 4; ++j) {
            const int m0 = mBase + wm * 32 + i * 16 + mh;
            const int n0 = nBase + wn * 64 + j * 16 + cn;
            #pragma unroll
            for (int r = 0; r < 8; ++r)
                out[(size_t)(m0 + r) * N_DIM + n0] = acc[i][j][r];
        }
    }
}

// ---------------- Fallback: fused conversion GEMM (round-1 kernel) ----------

__global__ __launch_bounds__(512)
void gemm_fused(const float* __restrict__ x,
                const float* __restrict__ W,
                const unsigned char* __restrict__ mask,
                float* __restrict__ out) {
    __shared__ __align__(16) unsigned short Ahi[128][LDSK];
    __shared__ __align__(16) unsigned short Alo[128][LDSK];
    __shared__ __align__(16) unsigned short Bhi[128][LDSK];
    __shared__ __align__(16) unsigned short Blo[128][LDSK];

    const int tid  = threadIdx.x;
    const int lane = tid & 31;
    const int wave = tid >> 5;
    const int wm   = wave >> 2;
    const int wn   = wave & 3;

    const int mBase = blockIdx.y * 128;
    const int nBase = blockIdx.x * 128;

    const int ldr = tid >> 2;
    const int ldc = (tid & 3) * 8;

    const float scale = 1.0f / 0.9f;

    v8f acc[2][2];
    #pragma unroll
    for (int i = 0; i < 2; ++i)
        #pragma unroll
        for (int j = 0; j < 2; ++j)
            acc[i][j] = v8f{};

    for (int kk = 0; kk < K_DIM; kk += 32) {
        {
            const size_t off = (size_t)(mBase + ldr) * K_DIM + kk + ldc;
            float4 a0 = *reinterpret_cast<const float4*>(x + off);
            float4 a1 = *reinterpret_cast<const float4*>(x + off + 4);
            unsigned long long mm =
                *reinterpret_cast<const unsigned long long*>(mask + off);
            float av[8] = {a0.x, a0.y, a0.z, a0.w, a1.x, a1.y, a1.z, a1.w};
            ushort8 h, l;
            #pragma unroll
            for (int e = 0; e < 8; ++e) {
                float f = ((mm >> (8 * e)) & 0xFFull) ? av[e] * scale : 0.0f;
                unsigned short hb = f32_to_bf16_rne(f);
                h[e] = hb;
                l[e] = f32_to_bf16_rne(f - bf16u_to_f32(hb));
            }
            *reinterpret_cast<ushort8*>(&Ahi[ldr][ldc]) = h;
            *reinterpret_cast<ushort8*>(&Alo[ldr][ldc]) = l;
        }
        {
            const size_t off = (size_t)(nBase + ldr) * K_DIM + kk + ldc;
            float4 b0 = *reinterpret_cast<const float4*>(W + off);
            float4 b1 = *reinterpret_cast<const float4*>(W + off + 4);
            float bv[8] = {b0.x, b0.y, b0.z, b0.w, b1.x, b1.y, b1.z, b1.w};
            ushort8 h, l;
            #pragma unroll
            for (int e = 0; e < 8; ++e) {
                float f = bv[e];
                unsigned short hb = f32_to_bf16_rne(f);
                h[e] = hb;
                l[e] = f32_to_bf16_rne(f - bf16u_to_f32(hb));
            }
            *reinterpret_cast<ushort8*>(&Bhi[ldr][ldc]) = h;
            *reinterpret_cast<ushort8*>(&Blo[ldr][ldc]) = l;
        }
        __syncthreads();

        v16bf aH[2], aL[2], bH[2], bL[2];
        #pragma unroll
        for (int i = 0; i < 2; ++i) {
            aH[i] = load_fragA(Ahi, wm * 32 + i * 16, lane);
            aL[i] = load_fragA(Alo, wm * 32 + i * 16, lane);
            bH[i] = load_fragB(Bhi, wn * 32 + i * 16, lane);
            bL[i] = load_fragB(Blo, wn * 32 + i * 16, lane);
        }

        #pragma unroll
        for (int i = 0; i < 2; ++i) {
            #pragma unroll
            for (int j = 0; j < 2; ++j) {
                v8f c = acc[i][j];
                c = __builtin_amdgcn_wmma_f32_16x16x32_bf16(
                        false, aL[i], false, bH[j], (short)0, c, false, false);
                c = __builtin_amdgcn_wmma_f32_16x16x32_bf16(
                        false, aH[i], false, bL[j], (short)0, c, false, false);
                c = __builtin_amdgcn_wmma_f32_16x16x32_bf16(
                        false, aH[i], false, bH[j], (short)0, c, false, false);
                acc[i][j] = c;
            }
        }
        __syncthreads();
    }

    const int cn = lane & 15;
    const int mh = (lane >> 4) * 8;
    #pragma unroll
    for (int i = 0; i < 2; ++i) {
        #pragma unroll
        for (int j = 0; j < 2; ++j) {
            const int m0 = mBase + wm * 32 + i * 16 + mh;
            const int n0 = nBase + wn * 32 + j * 16 + cn;
            #pragma unroll
            for (int r = 0; r < 8; ++r)
                out[(size_t)(m0 + r) * N_DIM + n0] = acc[i][j][r];
        }
    }
}

// ---------------- Host launcher ----------------

extern "C" void kernel_launch(void* const* d_in, const int* in_sizes, int n_in,
                              void* d_out, int out_size, void* d_ws, size_t ws_size,
                              hipStream_t stream) {
    (void)in_sizes; (void)n_in; (void)out_size;
    const float*         x    = (const float*)d_in[0];
    const float*         W    = (const float*)d_in[1];
    const unsigned char* mask = (const unsigned char*)d_in[2];
    float*               out  = (float*)d_out;

    const size_t elemsX = (size_t)T_DIM * K_DIM;   // 16.7M
    const size_t elemsW = (size_t)N_DIM * K_DIM;   // 16.7M
    const size_t bytesPerArr = elemsX * sizeof(unsigned short); // 32 MB each
    const size_t need = 4 * bytesPerArr;           // 128 MB

    dim3 grid(N_DIM / 128, T_DIM / 128);           // 64 x 64 blocks

    if (ws_size >= need) {
        unsigned char* base = (unsigned char*)d_ws;
        unsigned short* Ahi = (unsigned short*)(base + 0 * bytesPerArr);
        unsigned short* Alo = (unsigned short*)(base + 1 * bytesPerArr);
        unsigned short* Bhi = (unsigned short*)(base + 2 * bytesPerArr);
        unsigned short* Blo = (unsigned short*)(base + 3 * bytesPerArr);

        split_x_dropout<<<dim3((unsigned)(elemsX / 8 / 256)), 256, 0, stream>>>(
            x, mask, (ushort8*)Ahi, (ushort8*)Alo);
        split_w<<<dim3((unsigned)(elemsW / 8 / 256)), 256, 0, stream>>>(
            W, (ushort8*)Bhi, (ushort8*)Blo);
        gemm_bf16split<<<grid, 256, 0, stream>>>(Ahi, Alo, Bhi, Blo, out);
    } else {
        gemm_fused<<<grid, 512, 0, stream>>>(x, W, mask, out);
    }
}
